// RBN_28656021799451
// MI455X (gfx1250) — matrix operand order repaired and checked by reading, not compiled, not run on hardware
//
#include <hip/hip_runtime.h>
#include <hip/hip_bf16.h>
#include <math.h>

#define B_   8
#define C_   2048
#define D_   128
#define L_   3
#define H_   4
#define K_   200
#define S_   64
#define OUT_ 24
#define dh   32
#define KP   224          // K padded to multiple of 32 for WMMA tiling
#define BH   (B_ * H_)
#define NB   (B_ * KP * D_)   // 229376 floats

typedef __attribute__((ext_vector_type(16))) __bf16 v16bf;
typedef __attribute__((ext_vector_type(8)))  float  v8f;
typedef __attribute__((ext_vector_type(4)))  float  f4;
typedef __attribute__((ext_vector_type(4)))  int    v4i;

#if defined(__has_builtin)
#if __has_builtin(__builtin_amdgcn_global_load_async_to_lds_b128) && \
    __has_builtin(__builtin_amdgcn_s_wait_asynccnt)
#define USE_ASYNC_LDS 1
typedef __attribute__((address_space(1))) v4i gv4i;
typedef __attribute__((address_space(3))) v4i lv4i;
#endif
#endif

__device__ __forceinline__ float gelu_exact(float x) {
  return 0.5f * x * (1.0f + erff(x * 0.7071067811865476f));
}

// ---------------------------------------------------------------------------
// Kernel 1: per-batch-row top-K via bitonic sort in LDS, then role*filler bind
// (attention/readout are permutation invariant over slots -> order-free top-k)
// ---------------------------------------------------------------------------
__global__ void bind_topk_kernel(const float* __restrict__ x,
                                 const float* __restrict__ role_emb,
                                 const float* __restrict__ filler_w,
                                 float* __restrict__ bout) {
  __shared__ float key[C_];
  __shared__ int   kid[C_];
  const int b = blockIdx.x, t = threadIdx.x;
  for (int i = t; i < C_; i += 256) {
    float xv = x[b * C_ + i];
    key[i] = log1pf(fmaxf(xv, 0.0f));
    kid[i] = i;
  }
  __syncthreads();
  for (int kk = 2; kk <= C_; kk <<= 1) {
    for (int j = kk >> 1; j > 0; j >>= 1) {
      for (int i = t; i < C_; i += 256) {
        int ixj = i ^ j;
        if (ixj > i) {
          bool up = ((i & kk) == 0);
          float a = key[i], c = key[ixj];
          bool sw = up ? (a < c) : (a > c);
          if (sw) {
            key[i] = c; key[ixj] = a;
            int tmp = kid[i]; kid[i] = kid[ixj]; kid[ixj] = tmp;
          }
        }
      }
      __syncthreads();
    }
  }
  for (int e = t; e < KP * D_; e += 256) {
    int k = e / D_, dd = e % D_;
    float v = 0.0f;
    if (k < K_) {
      float val = key[k];
      int   id  = kid[k];
      float fil = gelu_exact(val * filler_w[dd]);
      v = role_emb[id * D_ + dd] * fil;
    }
    bout[(long)b * KP * D_ + e] = v;
  }
}

// ---------------------------------------------------------------------------
// Generic batched GEMM: C[z] = act(A[z] @ W[z] + bias), bf16 WMMA, f32 accum.
// One wave computes a 16 x (16*NT) tile: NT accumulators share one A fragment.
// A loads are 16B-aligned b128; W is loaded per element. M mult of 16, N mult of 16*NT, Kd mult of 32.
// ---------------------------------------------------------------------------
template <int NT, bool HB, bool ACT_GELU>
__global__ void gemm_wmma(const float* __restrict__ A,
                          const float* __restrict__ W,
                          const float* __restrict__ bias,
                          float* __restrict__ Cmat,
                          int Kd, int lda, int ldw, int ldc,
                          long long sA, long long sW, long long sC) {
  const int lane = threadIdx.x;       // 0..31
  const int mt = blockIdx.x, nt0 = blockIdx.y * NT, z = blockIdx.z;
  A    += (long long)z * sA;
  W    += (long long)z * sW;
  Cmat += (long long)z * sC;
  const int half = lane >> 4;
  const int ml   = lane & 15;

  v8f zero = {};
  v8f acc[NT];
#pragma unroll
  for (int u = 0; u < NT; ++u) acc[u] = zero;

  for (int kc = 0; kc < Kd; kc += 32) {
    // A fragment: lane (m = lane&15) needs K spans [8h..8h+7] and [16+8h..16+8h+7]
    const float* Arow = A + (long long)(mt * 16 + ml) * lda + kc + 8 * half;
    __builtin_prefetch(Arow + 32, 0, 1);
    f4 a0 = *(const f4*)(Arow);
    f4 a1 = *(const f4*)(Arow + 4);
    f4 a2 = *(const f4*)(Arow + 16);
    f4 a3 = *(const f4*)(Arow + 20);
    v16bf af;
#pragma unroll
    for (int s = 0; s < 4; ++s) {
      af[s]      = (__bf16)a0[s];
      af[4 + s]  = (__bf16)a1[s];
      af[8 + s]  = (__bf16)a2[s];
      af[12 + s] = (__bf16)a3[s];
    }
    // B fragments: lane n = lane&15, K spans [8h..8h+7] and [16+8h..16+8h+7]
    const float* Wcol = W + (long long)(kc + 8 * half) * ldw + nt0 * 16 + ml;
#pragma unroll
    for (int u = 0; u < NT; ++u) {
      v16bf bf;
#pragma unroll
      for (int s = 0; s < 8; ++s) {
        bf[s]     = (__bf16)Wcol[(long long)s * ldw + u * 16];
        bf[8 + s] = (__bf16)Wcol[(long long)(16 + s) * ldw + u * 16];
      }
      acc[u] = __builtin_amdgcn_wmma_f32_16x16x32_bf16(false, af, false, bf,
                                                       (short)0, acc[u], false, false);
    }
  }
#pragma unroll
  for (int u = 0; u < NT; ++u) {
    const int n = (nt0 + u) * 16 + ml;
    const float bv = HB ? bias[n] : 0.0f;
#pragma unroll
    for (int r = 0; r < 8; ++r) {
      int mrow = mt * 16 + r + 8 * half;
      float v = acc[u][r] + bv;
      if (ACT_GELU) v = gelu_exact(v);
      Cmat[(long long)mrow * ldc + n] = v;
    }
  }
}

// ---------------------------------------------------------------------------
// Head split / merge reshapes:  [B,KP,H*dh] <-> [B*H,KP,dh]
// ---------------------------------------------------------------------------
__global__ void split_heads(const float* __restrict__ in, float* __restrict__ out) {
  int idx = blockIdx.x * 256 + threadIdx.x;
  if (idx >= NB) return;
  int e = idx & 31;
  int tmp = idx >> 5;
  int i = tmp % KP; tmp /= KP;
  int h = tmp % H_; int b = tmp / H_;
  out[idx] = in[((long)b * KP + i) * D_ + h * dh + e];
}

__global__ void merge_heads(const float* __restrict__ in, float* __restrict__ out) {
  int idx = blockIdx.x * 256 + threadIdx.x;
  if (idx >= NB) return;
  int col = idx & 127;
  int row = idx >> 7;
  int i = row % KP, b = row / KP;
  int h = col >> 5, e = col & 31;
  out[idx] = in[(((long)b * H_ + h) * KP + i) * dh + e];
}

// ---------------------------------------------------------------------------
// Relational scores, fully fused. One wave per (z=b*H+h, 16i-tile, 16j-tile).
// itm[:,:,f] = (Q * W1i[:,f]) @ K^T  -> one 16x16x32 WMMA per f (32 total),
// then score += gelu(qt + kt + itm + b1) * w2 fused on the D fragment.
// ---------------------------------------------------------------------------
__global__ void rel_scores_kernel(const float* __restrict__ qh,
                                  const float* __restrict__ kh,
                                  const float* __restrict__ qt,
                                  const float* __restrict__ kt,
                                  const float* __restrict__ W1i,
                                  const float* __restrict__ b1,
                                  const float* __restrict__ w2,
                                  const float* __restrict__ b2p,
                                  float* __restrict__ scores) {
  __shared__ float qtS[16 * 32], ktS[16 * 32], w1iS[32 * 32], b1S[32], w2S[32];
  const int lane = threadIdx.x;
  const int it = blockIdx.x, jt = blockIdx.y, z = blockIdx.z;
  const float* qhz = qh + (long long)z * KP * dh;
  const float* khz = kh + (long long)z * KP * dh;
  const float* qtz = qt + (long long)z * KP * dh;
  const float* ktz = kt + (long long)z * KP * dh;

#if defined(USE_ASYNC_LDS)
  {
    // async DMA of the 4KB W1i block into LDS (8 iters x 32 lanes x b128)
    gv4i* gp = (gv4i*)(W1i);
    lv4i* lp = (lv4i*)(w1iS);
#pragma unroll
    for (int q8 = 0; q8 < 8; ++q8)
      __builtin_amdgcn_global_load_async_to_lds_b128(gp + q8 * 32 + lane,
                                                     lp + q8 * 32 + lane, 0, 0);
  }
#else
  for (int i2 = lane * 4; i2 < 1024; i2 += 128)
    *(f4*)(w1iS + i2) = *(const f4*)(W1i + i2);
#endif
  for (int i2 = lane * 4; i2 < 512; i2 += 128) {
    *(f4*)(qtS + i2) = *(const f4*)(qtz + it * 16 * 32 + i2);
    *(f4*)(ktS + i2) = *(const f4*)(ktz + jt * 16 * 32 + i2);
  }
  b1S[lane] = b1[lane];
  w2S[lane] = w2[lane];
#if defined(USE_ASYNC_LDS)
  __builtin_amdgcn_s_wait_asynccnt(0);
#endif
  __syncthreads();

  const int half = lane >> 4, ml = lane & 15;
  // B fragment: K^T tile, lane j = lane&15, e spans [8h..8h+7] and [16+8h..16+8h+7]
  v16bf bfrag;
  const float* krow = khz + (jt * 16 + ml) * 32 + 8 * half;
#pragma unroll
  for (int r = 0; r < 8; ++r) {
    bfrag[r]     = (__bf16)krow[r];
    bfrag[8 + r] = (__bf16)krow[16 + r];
  }
  // Unscaled Q fragment values (two contiguous 8-float spans) + e-indices
  float qa[16];
  const float* qrow = qhz + (it * 16 + ml) * 32 + 8 * half;
  {
    f4 a0 = *(const f4*)(qrow);
    f4 a1 = *(const f4*)(qrow + 4);
    f4 a2 = *(const f4*)(qrow + 16);
    f4 a3 = *(const f4*)(qrow + 20);
#pragma unroll
    for (int s = 0; s < 4; ++s) {
      qa[s] = a0[s]; qa[4 + s] = a1[s]; qa[8 + s] = a2[s]; qa[12 + s] = a3[s];
    }
  }
  int es[16];
#pragma unroll
  for (int s = 0; s < 16; ++s) {
    int r = s >> 1, p = s & 1;
    es[s] = ((r < 4) ? (2 * r) : (16 + 2 * (r - 4))) + 8 * half + p;
  }

  float sacc[8];
#pragma unroll
  for (int r = 0; r < 8; ++r) sacc[r] = 0.0f;

  for (int f = 0; f < 32; ++f) {
    v16bf af;
#pragma unroll
    for (int s = 0; s < 16; ++s)
      af[s] = (__bf16)(qa[s] * w1iS[es[s] * 32 + f]);
    v8f zero = {};
    v8f itm = __builtin_amdgcn_wmma_f32_16x16x32_bf16(false, af, false, bfrag,
                                                      (short)0, zero, false, false);
#pragma unroll
    for (int r = 0; r < 8; ++r) {
      int il = r + 8 * half;
      float tv = qtS[il * 32 + f] + ktS[ml * 32 + f] + b1S[f] + itm[r];
      sacc[r] += gelu_exact(tv) * w2S[f];
    }
  }
  const float b2v = *b2p;
#pragma unroll
  for (int r = 0; r < 8; ++r) {
    int il = r + 8 * half;
    scores[((long long)z * KP + it * 16 + il) * KP + jt * 16 + ml] =
        (sacc[r] + b2v) * 0.17677669529663687f; // 1/sqrt(32)
  }
}

// ---------------------------------------------------------------------------
// Row softmax over j < K_, in place; padded cols [K_,KP) zeroed.
// ---------------------------------------------------------------------------
__global__ void softmax_rows(float* __restrict__ scores) {
  __shared__ float red[256];
  const long long row = blockIdx.x;
  float* rp = scores + row * KP;
  const int t = threadIdx.x;
  float mx = -1e30f;
  for (int j = t; j < K_; j += 256) mx = fmaxf(mx, rp[j]);
  red[t] = mx; __syncthreads();
  for (int s = 128; s > 0; s >>= 1) { if (t < s) red[t] = fmaxf(red[t], red[t + s]); __syncthreads(); }
  mx = red[0]; __syncthreads();
  float sm = 0.0f;
  for (int j = t; j < K_; j += 256) sm += expf(rp[j] - mx);
  red[t] = sm; __syncthreads();
  for (int s = 128; s > 0; s >>= 1) { if (t < s) red[t] += red[t + s]; __syncthreads(); }
  const float inv = 1.0f / red[0]; __syncthreads();
  for (int j = t; j < KP; j += 256)
    rp[j] = (j < K_) ? expf(rp[j] - mx) * inv : 0.0f;
}

// ---------------------------------------------------------------------------
// b = LayerNorm(b + x) row-wise (D_=128, 128 threads per row)
// ---------------------------------------------------------------------------
__global__ void add_ln(float* __restrict__ bres, const float* __restrict__ xin,
                       const float* __restrict__ g, const float* __restrict__ bb) {
  __shared__ float red[128];
  const int row = blockIdx.x, t = threadIdx.x;
  float v = bres[(long long)row * D_ + t] + xin[(long long)row * D_ + t];
  red[t] = v; __syncthreads();
  for (int s = 64; s > 0; s >>= 1) { if (t < s) red[t] += red[t + s]; __syncthreads(); }
  const float mean = red[0] * (1.0f / D_); __syncthreads();
  const float dv = v - mean;
  red[t] = dv * dv; __syncthreads();
  for (int s = 64; s > 0; s >>= 1) { if (t < s) red[t] += red[t + s]; __syncthreads(); }
  const float var = red[0] * (1.0f / D_); __syncthreads();
  bres[(long long)row * D_ + t] = dv * rsqrtf(var + 1e-5f) * g[t] + bb[t];
}

// ---------------------------------------------------------------------------
// Final readout: task-query attention + slot memory + LN + head. One block/batch.
// ---------------------------------------------------------------------------
__global__ void readout_kernel(const float* __restrict__ bfin,
                               const float* __restrict__ slots,
                               const float* __restrict__ memg,
                               const float* __restrict__ memb,
                               const float* __restrict__ task_q,
                               const float* __restrict__ headW,
                               const float* __restrict__ headb,
                               float* __restrict__ out) {
  __shared__ float sc[K_], red[256], hv[D_], mv[D_], av[D_];
  const int b = blockIdx.x, t = threadIdx.x;
  const float* Bb = bfin + (long long)b * KP * D_;
  const float invsD = 0.08838834764831845f; // 1/sqrt(128)

  for (int c = t; c < K_; c += 256) {
    float d = 0.0f;
    const float* br = Bb + c * D_;
    for (int e = 0; e < D_; ++e) d += br[e] * task_q[e];
    sc[c] = d * invsD;
  }
  __syncthreads();
  {
    float m1 = -1e30f;
    for (int c = t; c < K_; c += 256) m1 = fmaxf(m1, sc[c]);
    red[t] = m1; __syncthreads();
    for (int s = 128; s > 0; s >>= 1) { if (t < s) red[t] = fmaxf(red[t], red[t + s]); __syncthreads(); }
    float mx = red[0]; __syncthreads();
    float s1 = 0.0f;
    for (int c = t; c < K_; c += 256) { float e = expf(sc[c] - mx); sc[c] = e; s1 += e; }
    red[t] = s1; __syncthreads();
    for (int s = 128; s > 0; s >>= 1) { if (t < s) red[t] += red[t + s]; __syncthreads(); }
    float inv = 1.0f / red[0]; __syncthreads();
    for (int c = t; c < K_; c += 256) sc[c] *= inv;
    __syncthreads();
  }
  if (t < D_) {
    float a = 0.0f;
    for (int c = 0; c < K_; ++c) a += sc[c] * Bb[c * D_ + t];
    hv[t] = a;
    mv[t] = 0.0f;
  }
  __syncthreads();

  for (int s2 = 0; s2 < S_; ++s2) {
    const float* sw = slots + s2 * D_;
    for (int c = t; c < K_; c += 256) {
      float d = 0.0f;
      const float* br = Bb + c * D_;
      for (int e = 0; e < D_; ++e) d += sw[e] * br[e];
      sc[c] = d * invsD;
    }
    __syncthreads();
    float m1 = -1e30f;
    for (int c = t; c < K_; c += 256) m1 = fmaxf(m1, sc[c]);
    red[t] = m1; __syncthreads();
    for (int s = 128; s > 0; s >>= 1) { if (t < s) red[t] = fmaxf(red[t], red[t + s]); __syncthreads(); }
    float mx = red[0]; __syncthreads();
    float s1 = 0.0f;
    for (int c = t; c < K_; c += 256) { float e = expf(sc[c] - mx); sc[c] = e; s1 += e; }
    red[t] = s1; __syncthreads();
    for (int s = 128; s > 0; s >>= 1) { if (t < s) red[t] += red[t + s]; __syncthreads(); }
    float inv = 1.0f / red[0]; __syncthreads();
    for (int c = t; c < K_; c += 256) sc[c] *= inv;
    __syncthreads();
    if (t < D_) {
      float a = 0.0f;
      for (int c = 0; c < K_; ++c) a += sc[c] * Bb[c * D_ + t];
      mv[t] += a;
    }
    __syncthreads();
  }

  if (t < D_) av[t] = mv[t] * (1.0f / S_);
  __syncthreads();
  red[t] = (t < D_) ? av[t] : 0.0f; __syncthreads();
  for (int s = 128; s > 0; s >>= 1) { if (t < s) red[t] += red[t + s]; __syncthreads(); }
  float mean = red[0] * (1.0f / D_); __syncthreads();
  red[t] = (t < D_) ? (av[t] - mean) * (av[t] - mean) : 0.0f; __syncthreads();
  for (int s = 128; s > 0; s >>= 1) { if (t < s) red[t] += red[t + s]; __syncthreads(); }
  float var = red[0] * (1.0f / D_); __syncthreads();
  if (t < D_)
    av[t] = (av[t] - mean) * rsqrtf(var + 1e-5f) * memg[t] + memb[t] + hv[t];
  __syncthreads();

  if (t < OUT_) {
    float o = headb[t];
    for (int e = 0; e < D_; ++e) o += av[e] * headW[e * OUT_ + t];
    out[b * OUT_ + t] = o;
  }
}

// ---------------------------------------------------------------------------
// Host side
// ---------------------------------------------------------------------------
static inline void launch_gemm(const float* A, const float* W, const float* bias,
                               float* Cm, int M, int N, int Kd,
                               int lda, int ldw, int ldc, int batch,
                               long long sA, long long sW, long long sC,
                               bool gelu, hipStream_t st) {
  if ((N % 64) == 0) {
    dim3 grid(M / 16, N / 64, batch);
    if (bias) {
      if (gelu)
        gemm_wmma<4, true, true><<<grid, dim3(32), 0, st>>>(A, W, bias, Cm, Kd, lda, ldw, ldc, sA, sW, sC);
      else
        gemm_wmma<4, true, false><<<grid, dim3(32), 0, st>>>(A, W, bias, Cm, Kd, lda, ldw, ldc, sA, sW, sC);
    } else {
      gemm_wmma<4, false, false><<<grid, dim3(32), 0, st>>>(A, W, bias, Cm, Kd, lda, ldw, ldc, sA, sW, sC);
    }
  } else { // N multiple of 32 (the dh=32 GEMMs; all bias-free, no activation)
    dim3 grid(M / 16, N / 32, batch);
    gemm_wmma<2, false, false><<<grid, dim3(32), 0, st>>>(A, W, bias, Cm, Kd, lda, ldw, ldc, sA, sW, sC);
  }
}

extern "C" void kernel_launch(void* const* d_in, const int* in_sizes, int n_in,
                              void* d_out, int out_size, void* d_ws, size_t ws_size,
                              hipStream_t stream) {
  const float* x        = (const float*)d_in[0];
  const float* role_emb = (const float*)d_in[1];
  const float* filler_w = (const float*)d_in[2];
  const float* qW  = (const float*)d_in[3];
  const float* qb  = (const float*)d_in[4];
  const float* kW  = (const float*)d_in[5];
  const float* kb  = (const float*)d_in[6];
  const float* vW  = (const float*)d_in[7];
  const float* vb  = (const float*)d_in[8];
  const float* oW  = (const float*)d_in[9];
  const float* ob  = (const float*)d_in[10];
  const float* relW1 = (const float*)d_in[11];
  const float* relb1 = (const float*)d_in[12];
  const float* relW2 = (const float*)d_in[13];
  const float* relb2 = (const float*)d_in[14];
  const float* ln1_g = (const float*)d_in[15];
  const float* ln1_b = (const float*)d_in[16];
  const float* ln2_g = (const float*)d_in[17];
  const float* ln2_b = (const float*)d_in[18];
  const float* ffnW1 = (const float*)d_in[19];
  const float* ffnb1 = (const float*)d_in[20];
  const float* ffnW2 = (const float*)d_in[21];
  const float* ffnb2 = (const float*)d_in[22];
  const float* slots = (const float*)d_in[23];
  const float* memg  = (const float*)d_in[24];
  const float* membv = (const float*)d_in[25];
  const float* task_q = (const float*)d_in[26];
  const float* headW  = (const float*)d_in[27];
  const float* headb  = (const float*)d_in[28];
  (void)in_sizes; (void)n_in; (void)out_size; (void)ws_size;

  float* ws = (float*)d_ws;
  float* b_cur  = ws;             // NB
  float* bufA   = ws + 1 * NB;    // q flat / attn-merge / ffn2 out
  float* bufB   = ws + 2 * NB;    // k flat / o-proj out
  float* bufC   = ws + 3 * NB;    // v flat
  float* qh     = ws + 4 * NB;
  float* kh     = ws + 5 * NB;
  float* vh     = ws + 6 * NB;
  float* qtb    = ws + 7 * NB;
  float* ktb    = ws + 8 * NB;
  float* oh     = ws + 9 * NB;
  float* scores = ws + 10 * NB;   // BH*KP*KP = 7*NB
  float* ffnh   = ws + 17 * NB;   // B*KP*4D = 4*NB

  bind_topk_kernel<<<B_, 256, 0, stream>>>(x, role_emb, filler_w, b_cur);

  const int M = B_ * KP; // 1792
  for (int l = 0; l < L_; ++l) {
    const float* qWl = qW + (long long)l * D_ * D_;
    const float* kWl = kW + (long long)l * D_ * D_;
    const float* vWl = vW + (long long)l * D_ * D_;
    const float* oWl = oW + (long long)l * D_ * D_;
    const float* relW1l = relW1 + (long long)l * 3 * dh * dh;

    launch_gemm(b_cur, qWl, qb + l * D_, bufA, M, D_, D_, D_, D_, D_, 1, 0, 0, 0, false, stream);
    launch_gemm(b_cur, kWl, kb + l * D_, bufB, M, D_, D_, D_, D_, D_, 1, 0, 0, 0, false, stream);
    launch_gemm(b_cur, vWl, vb + l * D_, bufC, M, D_, D_, D_, D_, D_, 1, 0, 0, 0, false, stream);

    split_heads<<<NB / 256, 256, 0, stream>>>(bufA, qh);
    split_heads<<<NB / 256, 256, 0, stream>>>(bufB, kh);
    split_heads<<<NB / 256, 256, 0, stream>>>(bufC, vh);

    // qt = q @ W1q , kt = k @ W1k   (per (b,h))
    launch_gemm(qh, relW1l,           nullptr, qtb, KP, dh, dh, dh, dh, dh,
                BH, (long long)KP * dh, 0, (long long)KP * dh, false, stream);
    launch_gemm(kh, relW1l + dh * dh, nullptr, ktb, KP, dh, dh, dh, dh, dh,
                BH, (long long)KP * dh, 0, (long long)KP * dh, false, stream);

    rel_scores_kernel<<<dim3(KP / 16, KP / 16, BH), 32, 0, stream>>>(
        qh, kh, qtb, ktb, relW1l + 2 * dh * dh, relb1 + l * dh, relW2 + l * dh,
        relb2 + l, scores);

    softmax_rows<<<BH * KP, 256, 0, stream>>>(scores);

    // o = attn @ v  (per (b,h)); padded attn cols are zero
    launch_gemm(scores, vh, nullptr, oh, KP, dh, KP, KP, dh, dh,
                BH, (long long)KP * KP, (long long)KP * dh, (long long)KP * dh, false, stream);

    merge_heads<<<NB / 256, 256, 0, stream>>>(oh, bufA);
    launch_gemm(bufA, oWl, ob + l * D_, bufB, M, D_, D_, D_, D_, D_, 1, 0, 0, 0, false, stream);
    add_ln<<<M, D_, 0, stream>>>(b_cur, bufB, ln1_g + l * D_, ln1_b + l * D_);

    launch_gemm(b_cur, ffnW1 + (long long)l * D_ * 4 * D_, ffnb1 + l * 4 * D_,
                ffnh, M, 4 * D_, D_, D_, 4 * D_, 4 * D_, 1, 0, 0, 0, true, stream);
    launch_gemm(ffnh, ffnW2 + (long long)l * 4 * D_ * D_, ffnb2 + l * D_,
                bufA, M, D_, 4 * D_, 4 * D_, D_, D_, 1, 0, 0, 0, false, stream);
    add_ln<<<M, D_, 0, stream>>>(b_cur, bufA, ln2_g + l * D_, ln2_b + l * D_);
  }

  readout_kernel<<<B_, 256, 0, stream>>>(b_cur, slots, memg, membv, task_q,
                                         headW, headb, (float*)d_out);
}
